// ConditionEncoder_81484119539722
// MI455X (gfx1250) — matrix operand order repaired and checked by reading, not compiled, not run on hardware
//
#include <hip/hip_runtime.h>

// ---------------------------------------------------------------------------
// CDNA5 (gfx1250) sparse-octree CNN as a chain of gather-GEMMs:
//  - v_wmma_f32_16x16x32_bf16 (bf16 in, f32 accumulate)
//  - global_load_async_to_lds_* (ASYNCcnt) for the 27-tap neighbor gather
//  - 32-row M tiles, K chunked by tap groups to keep LDS <= 48 KB
// ---------------------------------------------------------------------------

typedef __attribute__((ext_vector_type(16))) __bf16 v16bf;
typedef __attribute__((ext_vector_type(8)))  float  v8f;

union Frag32B {            // 32-byte fragment: 16 bf16 values, loaded as 2x16B
    uint4  q[2];
    v16bf  v;
};

__device__ __forceinline__ unsigned short f32_to_bf16(float f) {
    union { float f; unsigned int u; } c; c.f = f;
    unsigned int u = c.u;
    unsigned int r = u + 0x7FFFu + ((u >> 16) & 1u);   // round-to-nearest-even
    if ((u & 0x7F800000u) == 0x7F800000u) r = u;       // inf/nan passthrough
    return (unsigned short)(r >> 16);
}

__device__ __forceinline__ unsigned lds_offset_of(const void* p) {
    // generic-pointer low 32 bits == byte offset within the wave's LDS space
    return (unsigned)(size_t)p;
}

// ---------------------------------------------------------------------------
// f32 -> bf16 bulk convert (input features)
// ---------------------------------------------------------------------------
__global__ void convert_bf16_kernel(const float* __restrict__ in,
                                    unsigned short* __restrict__ out, int n) {
    for (int i = blockIdx.x * blockDim.x + threadIdx.x; i < n;
         i += gridDim.x * blockDim.x)
        out[i] = f32_to_bf16(in[i]);
}

// zero a small bf16 region (the async-gather zero page)
__global__ void zero_us_kernel(unsigned short* __restrict__ p, int n) {
    for (int i = blockIdx.x * blockDim.x + threadIdx.x; i < n;
         i += gridDim.x * blockDim.x)
        p[i] = 0;
}

// ---------------------------------------------------------------------------
// Pre-swizzle weights W[Ktot][Cout] (f32, K = 27*Cin flattened) into per-lane
// contiguous bf16 B-fragments: frag[(kb*nblks+nb)*32 + lane][16]
//   lane 0-15 : column n = nb*16+lane,    K = kb*32 + 0..15
//   lane 16-31: column n = nb*16+lane-16, K = kb*32 + 16..31
// ---------------------------------------------------------------------------
__global__ void prep_wfrag_kernel(const float* __restrict__ W,
                                  unsigned short* __restrict__ frag,
                                  int Ktot, int Cout, int kblks, int nblks) {
    int total = kblks * nblks * 32;
    for (int t = blockIdx.x * blockDim.x + threadIdx.x; t < total;
         t += gridDim.x * blockDim.x) {
        int lane = t & 31;
        int fid  = t >> 5;
        int kb   = fid / nblks;
        int nb   = fid - kb * nblks;
        int n    = nb * 16 + (lane & 15);
        int k0   = kb * 32 + ((lane >> 4) << 4);
        unsigned short* dst = frag + (size_t)t * 16;
#pragma unroll
        for (int j = 0; j < 16; ++j) {
            int k = k0 + j;
            float v = (k < Ktot && n < Cout) ? W[(size_t)k * Cout + n] : 0.0f;
            dst[j] = f32_to_bf16(v);
        }
    }
}

// ---------------------------------------------------------------------------
// Generic gather-GEMM sparse conv:
//   out[row, n] = relu?( sum_{tap,c} actIn[idx[tap,row], c] * W[tap,c,n] + b[n] )
// One block = 32 output rows (two 16-row WMMA tiles); wave w computes Cout
// columns [16w, 16w+16). K is processed in `nchunks` tap-group chunks; each
// chunk gathers 32 rows x TC taps into LDS via global_load_async_to_lds_*,
// then runs kcblks x (1 B-frag load + 2 WMMA) steps.
// ---------------------------------------------------------------------------
__global__ void gather_gemm_kernel(
    const unsigned short* __restrict__ actIn,   // Nin x Cin  (bf16)
    const unsigned short* __restrict__ zrow,    // 64 bf16 of zeros (128 B)
    const unsigned short* __restrict__ wfrag,   // (nchunks*kcblks)*nblks*32*16
    const float*          __restrict__ bias,    // Cout
    const int*            __restrict__ idx,     // [27][Nout]
    unsigned short*       __restrict__ outBf,   // Nout x Cout (bf16) or null
    float*                __restrict__ outF,    // Nout x Cout (f32)  or null
    int Nout, int Nin, int Cin, int Cout,
    int Ktot, int kcblks, int nchunks, int nblks, int relu)
{
    extern __shared__ __align__(16) unsigned char smem_raw[];
    unsigned short* lds = (unsigned short*)smem_raw;

    const int KcPad = kcblks * 32;              // K per chunk (padded)
    const int tid   = threadIdx.x;
    const int nthr  = blockDim.x;
    const int base  = blockIdx.x * 32;
    const unsigned ldsBase = lds_offset_of(lds);

    const int wave = tid >> 5;
    const int lane = tid & 31;
    const int m    = lane & 15;                 // A: lane -> row M
    const int hi   = lane >> 4;                 // upper lane half
    const int klo  = hi << 4;                   // A chunk K offset: 0 or 16

    v8f acc0 = {0.f, 0.f, 0.f, 0.f, 0.f, 0.f, 0.f, 0.f};
    v8f acc1 = {0.f, 0.f, 0.f, 0.f, 0.f, 0.f, 0.f, 0.f};

    const int TC = (Cin == 64) ? 9 : 27;        // taps per chunk

    for (int chunk = 0; chunk < nchunks; ++chunk) {
        if (chunk) __syncthreads();             // protect LDS before overwrite
        const int tapBase = chunk * TC;

        // ---- gather: 32 rows x TC taps, Cin bf16 each ---------------------
        if (Cin == 64) {
            // 8 x 16B async chunks per (row, tap)
            const int total = 32 * 9 * 8;
            for (int c = tid; c < total; c += nthr) {
                int r    = c / 72;              // 9*8
                int rem  = c - r * 72;
                int t    = rem >> 3;
                int q    = rem & 7;
                int row  = base + r;
                int tap  = tapBase + t;
                int src  = (row < Nout) ? idx[(size_t)tap * Nout + row] : Nin;
                unsigned long long g = (src < Nin)
                    ? (unsigned long long)(actIn + (size_t)src * 64 + q * 8)
                    : (unsigned long long)(zrow + q * 8);
                unsigned dlds =
                    ldsBase + (unsigned)((r * KcPad + t * 64 + q * 8) * 2);
                asm volatile("global_load_async_to_lds_b128 %0, %1, off"
                             :: "v"(dlds), "v"(g) : "memory");
            }
        } else if (Cin == 4) {
            // one 8B async chunk per (row, tap)
            const int total = 32 * 27;
            for (int p = tid; p < total; p += nthr) {
                int r   = p / 27;
                int tap = p - r * 27;
                int row = base + r;
                int src = (row < Nout) ? idx[(size_t)tap * Nout + row] : Nin;
                unsigned long long g = (src < Nin)
                    ? (unsigned long long)(actIn + (size_t)src * 4)
                    : (unsigned long long)zrow;
                unsigned dlds = ldsBase + (unsigned)((r * KcPad + tap * 4) * 2);
                asm volatile("global_load_async_to_lds_b64 %0, %1, off"
                             :: "v"(dlds), "v"(g) : "memory");
            }
        } else {
            // generic (Cin=27 pred2 layer): scalar copies through VGPRs
            const int total = 32 * 27;
            for (int p = tid; p < total; p += nthr) {
                int r   = p / 27;
                int tap = p - r * 27;
                int row = base + r;
                int src = (row < Nout) ? idx[(size_t)tap * Nout + row] : Nin;
                unsigned short* d = lds + (size_t)r * KcPad + tap * Cin;
                if (src < Nin) {
                    const unsigned short* s = actIn + (size_t)src * Cin;
                    for (int i = 0; i < Cin; ++i) d[i] = s[i];
                } else {
                    for (int i = 0; i < Cin; ++i) d[i] = 0;
                }
            }
        }
        // ---- zero the K padding of this chunk -----------------------------
        int kvalid = Ktot - chunk * KcPad;
        if (kvalid > KcPad) kvalid = KcPad;
        int padw = KcPad - kvalid;
        if (padw > 0) {
            for (int p = tid; p < 32 * padw; p += nthr) {
                int r = p / padw;
                int c = kvalid + (p - r * padw);
                lds[(size_t)r * KcPad + c] = 0;
            }
        }
        asm volatile("s_wait_asynccnt 0x0" ::: "memory");
        __syncthreads();

        // ---- WMMA K-loop: 1 B-frag -> 2 WMMAs (rows 0-15 and 16-31) -------
        const unsigned short* arow0 = lds + (size_t)m * KcPad;
        const unsigned short* arow1 = lds + (size_t)(16 + m) * KcPad;
        for (int kb = 0; kb < kcblks; ++kb) {
            int kbG = chunk * kcblks + kb;      // global K block for weights
            Frag32B a0, a1, b;
            const unsigned short* bp =
                wfrag + (((size_t)kbG * nblks + wave) * 32 + lane) * 16;
            b.q[0] = *(const uint4*)(bp);
            b.q[1] = *(const uint4*)(bp + 8);
            const unsigned short* ap0 = arow0 + kb * 32 + klo;
            a0.q[0] = *(const uint4*)(ap0);
            a0.q[1] = *(const uint4*)(ap0 + 8);
            const unsigned short* ap1 = arow1 + kb * 32 + klo;
            a1.q[0] = *(const uint4*)(ap1);
            a1.q[1] = *(const uint4*)(ap1 + 8);
            acc0 = __builtin_amdgcn_wmma_f32_16x16x32_bf16(
                false, a0.v, false, b.v, (short)0, acc0, false, false);
            acc1 = __builtin_amdgcn_wmma_f32_16x16x32_bf16(
                false, a1.v, false, b.v, (short)0, acc1, false, false);
        }
    }

    // ---- bias + relu + store (C/D layout: lane=N, vgpr v -> M=v+8*hi) -----
    const int ncol = wave * 16 + (lane & 15);
    const float bv = (ncol < Cout) ? bias[ncol] : 0.0f;
#pragma unroll
    for (int v = 0; v < 8; ++v) {
        int mrow = v + (hi << 3);
        float v0 = acc0[v] + bv;
        float v1 = acc1[v] + bv;
        if (relu) { v0 = fmaxf(v0, 0.0f); v1 = fmaxf(v1, 0.0f); }
        int r0 = base + mrow;
        int r1 = base + 16 + mrow;
        if (ncol < Cout) {
            if (r0 < Nout) {
                if (outBf) outBf[(size_t)r0 * Cout + ncol] = f32_to_bf16(v0);
                if (outF)  outF[(size_t)r0 * Cout + ncol]  = v0;
            }
            if (r1 < Nout) {
                if (outBf) outBf[(size_t)r1 * Cout + ncol] = f32_to_bf16(v1);
                if (outF)  outF[(size_t)r1 * Cout + ncol]  = v1;
            }
        }
    }
}

// ---------------------------------------------------------------------------
// Host-side orchestration
// ---------------------------------------------------------------------------
extern "C" void kernel_launch(void* const* d_in, const int* in_sizes, int n_in,
                              void* d_out, int out_size, void* d_ws, size_t ws_size,
                              hipStream_t stream) {
    (void)n_in; (void)out_size; (void)ws_size;

    const float* x       = (const float*)d_in[0];
    const float* pre_W   = (const float*)d_in[1];
    const float* pre_b   = (const float*)d_in[2];
    const float* down_W  = (const float*)d_in[3];
    const float* down_b  = (const float*)d_in[4];
    const float* conv1_W = (const float*)d_in[5];
    const float* conv1_b = (const float*)d_in[6];
    const float* conv2_W = (const float*)d_in[7];
    const float* conv2_b = (const float*)d_in[8];
    const float* pred1_W = (const float*)d_in[9];
    const float* pred1_b = (const float*)d_in[10];
    const float* pred2_W = (const float*)d_in[11];
    const float* pred2_b = (const float*)d_in[12];
    const int*   idx0    = (const int*)d_in[13];
    const int*   idx_d0  = (const int*)d_in[14];
    const int*   idx1    = (const int*)d_in[15];
    const int*   idx_d1  = (const int*)d_in[16];
    const int*   idx2    = (const int*)d_in[17];
    const int*   idx_d2  = (const int*)d_in[18];
    const int*   idx3    = (const int*)d_in[19];

    const int N0 = in_sizes[0] / 4;    // 256000
    const int N1 = in_sizes[14] / 27;  // 64000
    const int N2 = in_sizes[16] / 27;  // 16000
    const int N3 = in_sizes[18] / 27;  // 4000

    auto cdiv = [](int a, int b) { return (a + b - 1) / b; };

    // ---- carve workspace --------------------------------------------------
    size_t off = 0;
    auto alloc_us = [&](size_t elems) -> unsigned short* {
        unsigned short* p = (unsigned short*)((char*)d_ws + off);
        off = (off + elems * 2 + 255) & ~(size_t)255;
        return p;
    };

    unsigned short* zrow = alloc_us(64);            // 128 B zero page
    unsigned short* xbf  = alloc_us((size_t)N0 * 4);
    unsigned short* A0   = alloc_us((size_t)N0 * 64);
    unsigned short* Ad[3], *Ts[3], *Bs[3], *Ps[3];
    const int Ns[4] = {N0, N1, N2, N3};
    for (int i = 0; i < 3; ++i) {
        Ad[i] = alloc_us((size_t)Ns[i + 1] * 64);
        Ts[i] = alloc_us((size_t)Ns[i + 1] * 64);
        Bs[i] = alloc_us((size_t)Ns[i + 1] * 64);
        Ps[i] = alloc_us((size_t)Ns[i + 1] * 27);
    }

    struct WFrag { unsigned short* p; int kblks, nblks; };
    auto prep = [&](const float* W, int Ktot, int Cout) -> WFrag {
        int kb = cdiv(Ktot, 32), nb = cdiv(Cout, 16);
        // round K blocks up so chunked layers cover kcblks*nchunks exactly
        if (Ktot == 27 * 64) kb = 54;               // 3 chunks x 18
        WFrag f{alloc_us((size_t)kb * nb * 32 * 16), kb, nb};
        int total = kb * nb * 32;
        prep_wfrag_kernel<<<cdiv(total, 256), 256, 0, stream>>>(
            W, f.p, Ktot, Cout, kb, nb);
        return f;
    };

    zero_us_kernel<<<1, 64, 0, stream>>>(zrow, 64);
    convert_bf16_kernel<<<cdiv(N0 * 4, 256), 256, 0, stream>>>(x, xbf, N0 * 4);

    WFrag fPre = prep(pre_W, 27 * 4, 64);
    WFrag fDn[3], fC1[3], fC2[3], fP1[3], fP2[3];
    for (int i = 0; i < 3; ++i) {
        fDn[i] = prep(down_W  + (size_t)i * 27 * 64 * 64, 27 * 64, 64);
        fC1[i] = prep(conv1_W + (size_t)i * 27 * 64 * 64, 27 * 64, 64);
        fC2[i] = prep(conv2_W + (size_t)i * 27 * 64 * 64, 27 * 64, 64);
        fP1[i] = prep(pred1_W + (size_t)i * 27 * 64 * 27, 27 * 64, 27);
        fP2[i] = prep(pred2_W + (size_t)i * 27 * 27 * 27, 27 * 27, 27);
    }

    auto gemm = [&](const unsigned short* in, const WFrag& f, const float* bias,
                    const int* idx, unsigned short* oBf, float* oF,
                    int Nout, int Nin, int Cin, int Cout, int Ktot, int relu) {
        int nchunks = (Cin == 64) ? 3 : 1;
        int kcblks  = f.kblks / nchunks;            // 18 for Cin=64, else whole
        dim3 grd(cdiv(Nout, 32));
        dim3 blk(32 * f.nblks);
        size_t shmem = (size_t)32 * kcblks * 32 * 2;
        gather_gemm_kernel<<<grd, blk, shmem, stream>>>(
            in, zrow, f.p, bias, idx, oBf, oF, Nout, Nin, Cin, Cout,
            Ktot, kcblks, nchunks, f.nblks, relu);
    };

    // ---- output partitions: (Q, bg0, bg1, bg2) ---------------------------
    float* out = (float*)d_out;
    float* Qout = out;
    float* bg[3];
    bg[0] = out + (size_t)N3 * 64;
    bg[1] = bg[0] + (size_t)N1 * 27;
    bg[2] = bg[1] + (size_t)N2 * 27;

    // ---- network ----------------------------------------------------------
    gemm(xbf, fPre, pre_b, idx0, A0, nullptr, N0, N0, 4, 64, 27 * 4, 1);

    const int* idxd[3] = {idx_d0, idx_d1, idx_d2};
    const int* idxl[3] = {idx1, idx2, idx3};
    const unsigned short* Ain[3] = {A0, Bs[0], Bs[1]};

    for (int i = 0; i < 3; ++i) {
        int No = Ns[i + 1], Ni = Ns[i];
        gemm(Ain[i], fDn[i], down_b  + i * 64, idxd[i], Ad[i], nullptr,
             No, Ni, 64, 64, 27 * 64, 0);
        gemm(Ad[i],  fC1[i], conv1_b + i * 64, idxl[i], Ts[i], nullptr,
             No, No, 64, 64, 27 * 64, 1);
        gemm(Ts[i],  fC2[i], conv2_b + i * 64, idxl[i], Bs[i],
             (i == 2) ? Qout : nullptr, No, No, 64, 64, 27 * 64, 0);
        gemm(Bs[i],  fP1[i], pred1_b + i * 27, idxl[i], Ps[i], nullptr,
             No, No, 64, 27, 27 * 64, 1);
        gemm(Ps[i],  fP2[i], pred2_b + i * 27, idxl[i], nullptr, bg[i],
             No, No, 27, 27, 27 * 27, 0);
    }
}
